// kpts_decoder_temporal_76862734729725
// MI455X (gfx1250) — matrix operand order repaired and checked by reading, not compiled, not run on hardware
//
#include <hip/hip_runtime.h>
#include <hip/hip_bf16.h>

// ---- WMMA types (gfx1250, wave32) ----
typedef __attribute__((ext_vector_type(8)))  float  v8f;
typedef __attribute__((ext_vector_type(16))) __bf16 v16bf;
typedef __attribute__((ext_vector_type(4)))  int    v4i;

union Frag { v16bf v; uint4 q[2]; };

__device__ __forceinline__ unsigned short f2bf(float f) {
  unsigned int u = __float_as_uint(f);
  u += 0x7FFFu + ((u >> 16) & 1u);          // round-to-nearest-even
  return (unsigned short)(u >> 16);
}
__device__ __forceinline__ unsigned int pack2(float lo, float hi) {
  return (unsigned int)f2bf(lo) | ((unsigned int)f2bf(hi) << 16);
}

// ---- CDNA5 async global->LDS copy (ASYNCcnt), with synchronous fallback ----
#if __has_builtin(__builtin_amdgcn_global_load_async_to_lds_b128) && \
    __has_builtin(__builtin_amdgcn_s_wait_asynccnt)
#define HAS_ASYNC_LDS 1
#else
#define HAS_ASYNC_LDS 0
#endif

__device__ __forceinline__ void copy16_g2l(const void* g, void* l) {
#if HAS_ASYNC_LDS
  // param types per hipcc diagnostic: (AS1 v4i*, AS3 v4i*, int offset, int cpol)
  __builtin_amdgcn_global_load_async_to_lds_b128(
      (__attribute__((address_space(1))) v4i*)g,
      (__attribute__((address_space(3))) v4i*)l, 0, 0);
#else
  *(uint4*)l = *(const uint4*)g;
#endif
}
__device__ __forceinline__ void async_fence() {
#if HAS_ASYNC_LDS
  __builtin_amdgcn_s_wait_asynccnt(0);
#endif
}

// Bulk fp32 -> bf16 weight conversion (memory-bound pre-pass, runs once per call)
__global__ __launch_bounds__(256)
void convert_f32_bf16(const float* __restrict__ src, unsigned short* __restrict__ dst, int n4) {
  int i = blockIdx.x * 256 + threadIdx.x;
  if (i < n4) {
    float4 f = ((const float4*)src)[i];
    uint2 o;
    o.x = pack2(f.x, f.y);
    o.y = pack2(f.z, f.w);
    ((uint2*)dst)[i] = o;
  }
}

// One block: (M=128 batch) x (N=64 channels) tile of  out = gather(Hin) @ W^T + bias
//  GATHER=false: A = dense fp32 X (stage-0), converted on the fly
//  GATHER=true : A[m, l*CIN+c] = Hin[m, idx[node][l], c] (bf16) -> async DMA to LDS
//  WBF=true    : weights already bf16 in workspace -> async DMA to LDS
// Ping-pong LDS: async-stage chunk k+1 while WMMA consumes chunk k; the wave
// waits on its ASYNCcnt only after issuing the WMMAs (1 barrier/iter).
template<int KTOT, int CIN, int COUT, int NT, bool GATHER, bool WBF, bool DO_ELU, bool FINAL>
__global__ __launch_bounds__(256)
void fused_gemm(const float* __restrict__ X,
                const unsigned short* __restrict__ Hin,
                const float* __restrict__ Wf,            // fp32 weights (stage-0)
                const unsigned short* __restrict__ Wb,   // bf16 weights (layers)
                const float* __restrict__ Bias,
                const int* __restrict__ Idx,
                unsigned short* __restrict__ Hout,
                float* __restrict__ Fout)
{
  constexpr int BUF = 128 * 80 + 64 * 80;   // 15360 B per stage buffer
  __shared__ __align__(16) unsigned char smem[2 * BUF];
  __shared__ int sIdx[48];

  const int tid  = threadIdx.x;
  const int lane = tid & 31;
  const int wave = tid >> 5;
  const int ml   = lane & 15;
  const int hi   = lane >> 4;
  const int node    = GATHER ? ((int)blockIdx.x / NT) : 0;
  const int ch_base = GATHER ? ((int)blockIdx.x % NT) * 64 : (int)blockIdx.x * 64;
  const int mwb = (wave & 3) * 32;   // wave's M base (of 128)
  const int nwb = (wave >> 2) * 32;  // wave's N base (of 64)

  if constexpr (GATHER) {
    if (tid < 41) sIdx[tid] = Idx[node * 41 + tid];
  }
  __syncthreads();

  v8f acc[2][2];
  acc[0][0] = {}; acc[0][1] = {}; acc[1][0] = {}; acc[1][1] = {};

  // ---- staging: 128x32 A tile + 64x32 B tile (bf16, 80B padded rows) ----
  auto stage = [&](int kb, int sel) {
    unsigned char* ldsA = smem + sel * BUF;
    unsigned char* ldsB = ldsA + 128 * 80;
    {
      const int m = tid >> 1;
      const int part = tid & 1;
      if constexpr (GATHER) {
        const int j  = sIdx[kb / CIN];      // CIN % 32 == 0 -> chunk within one l
        const int cb = kb % CIN;
        const unsigned short* src = Hin + ((size_t)m * 80 + j) * CIN + cb + part * 16;
        unsigned char* dst = ldsA + m * 80 + part * 32;
        copy16_g2l(src, dst);
        copy16_g2l(src + 8, dst + 16);
      } else {
        const float4* src = (const float4*)(X + (size_t)m * KTOT + kb + part * 16);
        float4 f0 = src[0], f1 = src[1], f2 = src[2], f3 = src[3];
        uint4 o0, o1;
        o0.x = pack2(f0.x, f0.y); o0.y = pack2(f0.z, f0.w);
        o0.z = pack2(f1.x, f1.y); o0.w = pack2(f1.z, f1.w);
        o1.x = pack2(f2.x, f2.y); o1.y = pack2(f2.z, f2.w);
        o1.z = pack2(f3.x, f3.y); o1.w = pack2(f3.z, f3.w);
        uint4* dst = (uint4*)(ldsA + m * 80 + part * 32);
        dst[0] = o0;
        dst[1] = o1;
      }
    }
    {
      const int n = tid >> 2;
      const int seg = tid & 3;
      const int ch = ch_base + n;
      if constexpr (WBF && !FINAL) {
        copy16_g2l(Wb + (size_t)ch * KTOT + kb + seg * 8, ldsB + n * 80 + seg * 16);
      } else if constexpr (WBF) {          // FINAL: needs zero-fill for ch >= COUT
        uint4 z = {0u, 0u, 0u, 0u};
        uint4 oq = (ch < COUT) ? *(const uint4*)(Wb + (size_t)ch * KTOT + kb + seg * 8) : z;
        *(uint4*)(ldsB + n * 80 + seg * 16) = oq;
      } else {
        const float4* src = (const float4*)(Wf + (size_t)ch * KTOT + kb + seg * 8);
        float4 f0 = src[0], f1 = src[1];
        uint4 oq;
        oq.x = pack2(f0.x, f0.y); oq.y = pack2(f0.z, f0.w);
        oq.z = pack2(f1.x, f1.y); oq.w = pack2(f1.z, f1.w);
        *(uint4*)(ldsB + n * 80 + seg * 16) = oq;
      }
    }
  };

  // ---- compute: 4 WMMAs on one staged chunk ----
  auto compute = [&](int sel) {
    unsigned char* ldsA = smem + sel * BUF;
    unsigned char* ldsB = ldsA + 128 * 80;
    Frag a0, a1, b0, b1;
    a0.q[0] = *(const uint4*)(ldsA + (mwb + ml) * 80 + hi * 16);
    a0.q[1] = *(const uint4*)(ldsA + (mwb + ml) * 80 + 32 + hi * 16);
    a1.q[0] = *(const uint4*)(ldsA + (mwb + 16 + ml) * 80 + hi * 16);
    a1.q[1] = *(const uint4*)(ldsA + (mwb + 16 + ml) * 80 + 32 + hi * 16);
    b0.q[0] = *(const uint4*)(ldsB + (nwb + ml) * 80 + hi * 16);
    b0.q[1] = *(const uint4*)(ldsB + (nwb + ml) * 80 + 32 + hi * 16);
    b1.q[0] = *(const uint4*)(ldsB + (nwb + 16 + ml) * 80 + hi * 16);
    b1.q[1] = *(const uint4*)(ldsB + (nwb + 16 + ml) * 80 + 32 + hi * 16);
    acc[0][0] = __builtin_amdgcn_wmma_f32_16x16x32_bf16(false, a0.v, false, b0.v, (short)0, acc[0][0], false, false);
    acc[0][1] = __builtin_amdgcn_wmma_f32_16x16x32_bf16(false, a0.v, false, b1.v, (short)0, acc[0][1], false, false);
    acc[1][0] = __builtin_amdgcn_wmma_f32_16x16x32_bf16(false, a1.v, false, b0.v, (short)0, acc[1][0], false, false);
    acc[1][1] = __builtin_amdgcn_wmma_f32_16x16x32_bf16(false, a1.v, false, b1.v, (short)0, acc[1][1], false, false);
  };

  constexpr int NK = KTOT / 32;
  stage(0, 0);
  async_fence();
  __syncthreads();
  int p = 0;
  for (int ki = 0; ki < NK; ++ki) {
    if (ki + 1 < NK) stage((ki + 1) * 32, p ^ 1);   // async prefetch next chunk
    compute(p);                                      // WMMA on current chunk
    async_fence();                                   // drain own DMA after WMMAs issued
    __syncthreads();
    p ^= 1;
  }

  // ---- epilogue: bias (+ELU), store bf16 activations or final fp32 ----
  for (int mt = 0; mt < 2; ++mt) {
    for (int nt = 0; nt < 2; ++nt) {
      const int nl = nwb + nt * 16 + ml;
      const int ch = ch_base + nl;
      float bias;
      if constexpr (FINAL) bias = (ch < COUT) ? Bias[ch] : 0.0f;
      else                 bias = Bias[ch];
      for (int r = 0; r < 8; ++r) {
        const int M = mwb + mt * 16 + hi * 8 + r;  // C/D layout: M = r + 8*hi
        float v = acc[mt][nt][r] + bias;
        if (DO_ELU) v = (v > 0.0f) ? v : (__expf(v) - 1.0f);
        if constexpr (FINAL) {
          if (ch < COUT) Fout[(size_t)M * 160 + node * 2 + ch] = v;
        } else if constexpr (GATHER) {
          Hout[(size_t)M * (80 * COUT) + node * COUT + ch] = f2bf(v);
        } else {
          Hout[(size_t)M * 20480 + ch] = f2bf(v);
        }
      }
    }
  }
}

extern "C" void kernel_launch(void* const* d_in, const int* in_sizes, int n_in,
                              void* d_out, int out_size, void* d_ws, size_t ws_size,
                              hipStream_t stream) {
  const float* x   = (const float*)d_in[0];
  const float* W0  = (const float*)d_in[1];
  const float* b0  = (const float*)d_in[2];
  const float* W1  = (const float*)d_in[3];
  const float* b1  = (const float*)d_in[4];
  const float* W2  = (const float*)d_in[5];
  const float* b2  = (const float*)d_in[6];
  const float* W3  = (const float*)d_in[7];
  const float* b3  = (const float*)d_in[8];
  const float* W4  = (const float*)d_in[9];
  const float* b4  = (const float*)d_in[10];
  const int*   idx = (const int*)d_in[11];
  float* out = (float*)d_out;

  // workspace layout (bytes)
  char* ws = (char*)d_ws;
  unsigned short* bufA = (unsigned short*)(ws);              // 5,242,880 B (h0, h2)
  unsigned short* bufB = (unsigned short*)(ws + 5242880);    // 5,242,880 B (h1, h3)
  unsigned short* w1b  = (unsigned short*)(ws + 10485760);   // 256*10496*2 = 5,373,952 B
  unsigned short* w2b  = (unsigned short*)(ws + 15859712);   // 128*10496*2 = 2,686,976 B
  unsigned short* w3b  = (unsigned short*)(ws + 18546688);   //  64*5248*2  =   671,744 B
  unsigned short* w4b  = (unsigned short*)(ws + 19218432);   //   2*2624*2  =    10,496 B

  // pre-convert layer weights to bf16 (bandwidth-bound, ~1 us total)
  {
    const int n1 = 256 * 10496 / 4, n2 = 128 * 10496 / 4, n3 = 64 * 5248 / 4, n4 = 2 * 2624 / 4;
    convert_f32_bf16<<<(n1 + 255) / 256, 256, 0, stream>>>(W1, w1b, n1);
    convert_f32_bf16<<<(n2 + 255) / 256, 256, 0, stream>>>(W2, w2b, n2);
    convert_f32_bf16<<<(n3 + 255) / 256, 256, 0, stream>>>(W3, w3b, n3);
    convert_f32_bf16<<<(n4 + 255) / 256, 256, 0, stream>>>(W4, w4b, n4);
  }

  // stage 0: (128x1024) @ W0^T -> (128, 80, 256) bf16
  fused_gemm<1024, 1, 20480, 1, false, false, false, false>
      <<<320, 256, 0, stream>>>(x, nullptr, W0, nullptr, b0, nullptr, bufA, nullptr);
  // layer 1: K = 41*256, 256 -> 256, ELU
  fused_gemm<41 * 256, 256, 256, 4, true, true, true, false>
      <<<320, 256, 0, stream>>>(nullptr, bufA, nullptr, w1b, b1, idx, bufB, nullptr);
  // layer 2: K = 41*256, 256 -> 128, ELU
  fused_gemm<41 * 256, 256, 128, 2, true, true, true, false>
      <<<160, 256, 0, stream>>>(nullptr, bufB, nullptr, w2b, b2, idx, bufA, nullptr);
  // layer 3: K = 41*128, 128 -> 64, ELU
  fused_gemm<41 * 128, 128, 64, 1, true, true, true, false>
      <<<80, 256, 0, stream>>>(nullptr, bufA, nullptr, w3b, b3, idx, bufB, nullptr);
  // layer 4: K = 41*64, 64 -> 2, fp32 out
  fused_gemm<41 * 64, 64, 2, 1, true, true, false, true>
      <<<80, 256, 0, stream>>>(nullptr, bufB, nullptr, w4b, b4, idx, nullptr, out);
}
